// GaussianGPT_38104949850258
// MI455X (gfx1250) — matrix operand order repaired and checked by reading, not compiled, not run on hardware
//
#include <hip/hip_runtime.h>
#include <hip/hip_bf16.h>

#define L_  12
#define H_  12
#define C_  768
#define D_  64
#define FF_ 3072
#define B_  2
#define T_  2048
#define M_  (B_*T_)          // 4096 tokens
#define VP_ 4098
#define VF_ 4096

typedef __bf16 bf16;
typedef __attribute__((ext_vector_type(16))) __bf16 v16bf;
typedef __attribute__((ext_vector_type(8)))  __bf16 v8bf;
typedef __attribute__((ext_vector_type(8)))  float  v8f;

#if defined(__gfx1250__) && __has_builtin(__builtin_amdgcn_global_load_async_to_lds_b128) && __has_builtin(__builtin_amdgcn_s_wait_asynccnt)
#define HAVE_ASYNC 1
typedef int v4i_g __attribute__((vector_size(16)));           // int32x4 (gcc-style)
typedef __attribute__((address_space(1))) v4i_g* gasp;        // global int32x4*
typedef __attribute__((address_space(3))) v4i_g* lasp;        // LDS    int32x4*
#else
#define HAVE_ASYNC 0
#endif

static __device__ __forceinline__ v8f wmma_bf16(v16bf a, v16bf b, v8f c) {
  // D(16x16,f32) = A(16x32,bf16) x B(32x16,bf16) + C
  return __builtin_amdgcn_wmma_f32_16x16x32_bf16(false, a, false, b, (short)0, c,
                                                 false, false);
}

static __device__ __forceinline__ v16bf cat16(v8bf lo, v8bf hi) {
  v16bf r;
#pragma unroll
  for (int i = 0; i < 8; i++) { r[i] = lo[i]; r[8 + i] = hi[i]; }
  return r;
}

// ---------------------------------------------------------------- convert
__global__ __launch_bounds__(256) void k_f32_to_bf16(const float* __restrict__ in,
                                                     bf16* __restrict__ out, int n) {
  int i = blockIdx.x * 256 + threadIdx.x;
  if (i < n) out[i] = (bf16)in[i];
}

// ---------------------------------------------------------------- embedding
__global__ __launch_bounds__(256) void k_embed(const int* __restrict__ tokens,
                                               const int* __restrict__ ttype,
                                               const float* __restrict__ pos_emb,
                                               const float* __restrict__ feat_emb,
                                               float* __restrict__ x) {
  int idx = blockIdx.x * 256 + threadIdx.x;     // < M_*C_
  int m = idx / C_;
  int c = idx % C_;
  int tok = tokens[m];
  if (ttype[m] == 1) {
    int t = tok < 0 ? 0 : (tok > VF_ - 1 ? VF_ - 1 : tok);
    x[idx] = feat_emb[t * C_ + c];
  } else {
    int t = tok < 0 ? 0 : (tok > VP_ - 1 ? VP_ - 1 : tok);
    x[idx] = pos_emb[t * C_ + c];
  }
}

// ---------------------------------------------------------------- layernorm
// one wave (32 lanes) per row of C_=768; output bf16
__global__ __launch_bounds__(256) void k_ln(const float* x, const float* __restrict__ w,
                                            const float* __restrict__ b,
                                            bf16* __restrict__ out) {
  int wv = threadIdx.x >> 5, lane = threadIdx.x & 31;
  int row = blockIdx.x * 8 + wv;
  const float* xr = x + (long)row * C_;
  float s = 0.f, s2 = 0.f;
#pragma unroll
  for (int i = 0; i < C_ / 32; i++) {
    float v = xr[lane + i * 32];
    s += v; s2 += v * v;
  }
#pragma unroll
  for (int o = 16; o >= 1; o >>= 1) {
    s += __shfl_xor(s, o, 32);
    s2 += __shfl_xor(s2, o, 32);
  }
  float mean = s * (1.0f / C_);
  float var = s2 * (1.0f / C_) - mean * mean;
  float rstd = rsqrtf(var + 1e-5f);
#pragma unroll
  for (int i = 0; i < C_ / 32; i++) {
    int c = lane + i * 32;
    out[(long)row * C_ + c] = (bf16)((xr[c] - mean) * rstd * w[c] + b[c]);
  }
}

// ---------------------------------------------------------------- GEMM
// C[M,N] = A[M,K](bf16) x W[N,K]^T(bf16), f32 accumulate via WMMA.
// Per wave: 16x64 output strip (4 accumulators). A operand is staged into a
// per-wave double-buffered LDS tile via async global->LDS copies (ASYNCcnt),
// overlapping the copy of chunk i+1 with the WMMAs of chunk i. B columns are
// clamped (not branched) so the K-loop has no exec-mask divergence. Modes:
//   0: outf = acc (f32)
//   1: outf = cs[0]*resid + cr[0]*acc (f32, resid may alias outf)
//   2: outb = bf16(gelu_exact(acc))
__global__ __launch_bounds__(256) void k_gemm(const bf16* __restrict__ A,
                                              const bf16* __restrict__ W,
                                              float* outf, bf16* __restrict__ outb,
                                              const float* resid,
                                              const float* __restrict__ cr,
                                              const float* __restrict__ cs,
                                              int M, int N, int K, int mode) {
  __shared__ __align__(16) bf16 atile[8][2][512];   // per-wave 2 x (16x32) bf16
  int wv = threadIdx.x >> 5, lane = threadIdx.x & 31;
  int hf = lane >> 4, l15 = lane & 15;
  int nstrips = (N + 63) >> 6;
  int tw = (M >> 4) * nstrips;
  int wid = blockIdx.x * 8 + wv;
  if (wid >= tw) return;
  int mt = wid / nstrips, ns = wid % nstrips;
  int m0 = mt << 4, n0 = ns << 6;

  v8f acc[4];
#pragma unroll
  for (int j = 0; j < 4; j++)
#pragma unroll
    for (int i = 0; i < 8; i++) acc[j][i] = 0.f;

  // B base pointers with clamped columns (garbage columns are never stored)
  const bf16* wp[4];
#pragma unroll
  for (int j = 0; j < 4; j++) {
    int n = n0 + j * 16 + l15;
    if (n > N - 1) n = N - 1;
    wp[j] = W + (long)n * K + hf * 16;
  }

  bf16* lbuf = &atile[wv][0][0];
  // stage one 16x32 A chunk (1KB) into LDS buffer `buf`; 2 x 16B per lane
  auto stage = [&](int k0, int buf) {
#pragma unroll
    for (int pc = 0; pc < 2; pc++) {
      int oe = pc * 256 + lane * 8;          // element offset in tile [0,512)
      int r = oe >> 5, ce = oe & 31;         // row, col-element
      const bf16* src = A + (long)(m0 + r) * K + k0 + ce;
      bf16* dst = lbuf + buf * 512 + oe;
#if HAVE_ASYNC
      __builtin_amdgcn_global_load_async_to_lds_b128((gasp)src, (lasp)dst, 0, 0);
#else
      *(v8bf*)dst = *(const v8bf*)src;
#endif
    }
  };

  int nch = K >> 5;
  stage(0, 0);
  for (int i = 0; i < nch; ++i) {
    int cb = i & 1;
    if (i + 1 < nch) {
      stage((i + 1) << 5, cb ^ 1);
#if HAVE_ASYNC
      __builtin_amdgcn_s_wait_asynccnt(2);   // only next chunk's 2 copies pending
#endif
    } else {
#if HAVE_ASYNC
      __builtin_amdgcn_s_wait_asynccnt(0);   // drain before last compute
#endif
    }
    __asm__ __volatile__("" ::: "memory");
    if (i + 8 < nch) __builtin_prefetch(wp[0] + ((i + 8) << 5), 0, 1);
    const bf16* pa = lbuf + cb * 512 + l15 * 32 + hf * 16;
    v16bf af = cat16(*(const v8bf*)pa, *(const v8bf*)(pa + 8));
#pragma unroll
    for (int j = 0; j < 4; j++) {
      v16bf bw = *(const v16bf*)(wp[j] + (i << 5));
      acc[j] = wmma_bf16(af, bw, acc[j]);
    }
  }

  float rc = (mode == 1) ? cr[0] : 0.f;
  float sc = (mode == 1) ? cs[0] : 0.f;
#pragma unroll
  for (int j = 0; j < 4; j++) {
#pragma unroll
    for (int v = 0; v < 8; v++) {
      int m = m0 + v + 8 * hf;
      int n = n0 + j * 16 + l15;
      if (n >= N) continue;
      long idx = (long)m * N + n;
      float val = acc[j][v];
      if (mode == 0) {
        outf[idx] = val;
      } else if (mode == 1) {
        outf[idx] = sc * resid[idx] + rc * val;
      } else {
        float g = 0.5f * val * (1.0f + erff(val * 0.70710678118f));
        outb[idx] = (bf16)g;
      }
    }
  }
}

// ---------------------------------------------------------------- QKNorm + RoPE3D
// one wave per (b,h,t); lane handles dims 2*lane, 2*lane+1 (one rotation pair).
// writes q,k as (B*H, T, D) bf16 and v transposed as (B*H, D, T) bf16.
__global__ __launch_bounds__(256) void k_qkrope(const float* __restrict__ qkv,
                                                const int* __restrict__ coords,
                                                const float* __restrict__ qs,
                                                const float* __restrict__ ks,
                                                bf16* __restrict__ qo,
                                                bf16* __restrict__ ko,
                                                bf16* __restrict__ vto) {
  int wv = threadIdx.x >> 5, lane = threadIdx.x & 31;
  int bht = blockIdx.x * 8 + wv;           // < B_*H_*T_
  int t = bht % T_;
  int bh = bht / T_;
  int h = bh % H_;
  int b = bh / H_;
  const float* base = qkv + (long)(b * T_ + t) * (3 * C_) + h * D_;
  int d0 = lane * 2;
  float q0 = base[d0], q1 = base[d0 + 1];
  float k0 = base[C_ + d0], k1 = base[C_ + d0 + 1];
  float v0 = base[2 * C_ + d0], v1 = base[2 * C_ + d0 + 1];

  float sq = q0 * q0 + q1 * q1;
  float sk = k0 * k0 + k1 * k1;
#pragma unroll
  for (int o = 16; o >= 1; o >>= 1) {
    sq += __shfl_xor(sq, o, 32);
    sk += __shfl_xor(sk, o, 32);
  }
  float rq = rsqrtf(sq * (1.0f / D_) + 1e-6f);
  float rk = rsqrtf(sk * (1.0f / D_) + 1e-6f);
  q0 = q0 * rq * qs[d0]; q1 = q1 * rq * qs[d0 + 1];
  k0 = k0 * rk * ks[d0]; k1 = k1 * rk * ks[d0 + 1];

  int p = lane;                            // pair index 0..31; pairs 0..29 rotate
  if (p < 30) {
    int axis = p / 10, fi = p % 10;
    float freq = __powf(10000.f, -(float)fi * 0.1f);
    float ang = (float)coords[(long)(b * T_ + t) * 3 + axis] * freq;
    float cs_ = cosf(ang), sn_ = sinf(ang);
    float a = q0, bb = q1;
    q0 = a * cs_ - bb * sn_; q1 = a * sn_ + bb * cs_;
    a = k0; bb = k1;
    k0 = a * cs_ - bb * sn_; k1 = a * sn_ + bb * cs_;
  }
  long qi = ((long)bh * T_ + t) * D_ + d0;
  qo[qi] = (bf16)q0; qo[qi + 1] = (bf16)q1;
  ko[qi] = (bf16)k0; ko[qi + 1] = (bf16)k1;
  vto[((long)bh * D_ + d0) * T_ + t] = (bf16)v0;
  vto[((long)bh * D_ + d0 + 1) * T_ + t] = (bf16)v1;
}

// ---------------------------------------------------------------- flash attention
// one wave per (b,h, 16-query tile); online softmax over 32-key chunks.
__global__ __launch_bounds__(256) void k_attn(const bf16* __restrict__ qbf,
                                              const bf16* __restrict__ kbf,
                                              const bf16* __restrict__ vtbf,
                                              bf16* __restrict__ obf) {
  __shared__ __align__(16) bf16 pl[8 * 512];   // per-wave 16x32 P tile
  int wv = threadIdx.x >> 5, lane = threadIdx.x & 31;
  int hf = lane >> 4, l15 = lane & 15;
  int wid = blockIdx.x * 8 + wv;               // < B_*H_*(T_/16)
  int qt = wid % (T_ / 16);
  int bh = wid / (T_ / 16);
  int q0 = qt * 16;
  bf16* pw = pl + wv * 512;

  // load q A-fragments for the two K=32 steps of D=64
  v16bf aq[2];
#pragma unroll
  for (int kk = 0; kk < 2; kk++) {
    const bf16* ap = qbf + ((long)bh * T_ + q0 + l15) * D_ + kk * 32;
    aq[kk] = cat16(*(const v8bf*)(ap + hf * 16), *(const v8bf*)(ap + 8 + hf * 16));
  }

  v8f oacc[4];
#pragma unroll
  for (int j = 0; j < 4; j++)
#pragma unroll
    for (int i = 0; i < 8; i++) oacc[j][i] = 0.f;
  float mrow[8], lrow[8];
#pragma unroll
  for (int v = 0; v < 8; v++) { mrow[v] = -3.0e38f; lrow[v] = 0.f; }

  for (int c0 = 0; c0 <= q0 + 15; c0 += 32) {
    // scores: two 16x16 tiles (keys c0..c0+15, c0+16..c0+31), K = D = 64
    v8f s[2];
#pragma unroll
    for (int st = 0; st < 2; st++) {
      const bf16* kb = kbf + ((long)bh * T_ + c0 + st * 16 + l15) * D_;
      v16bf bk0 = *(const v16bf*)(kb + hf * 16);
      v16bf bk1 = *(const v16bf*)(kb + 32 + hf * 16);
      v8f sv;
#pragma unroll
      for (int i = 0; i < 8; i++) sv[i] = 0.f;
      sv = wmma_bf16(aq[0], bk0, sv);
      sv = wmma_bf16(aq[1], bk1, sv);
      s[st] = sv;
    }
    // scale + causal mask
#pragma unroll
    for (int st = 0; st < 2; st++) {
      int key = c0 + st * 16 + l15;
#pragma unroll
      for (int v = 0; v < 8; v++) {
        int row = q0 + v + 8 * hf;
        float val = s[st][v] * 0.125f;
        s[st][v] = (key > row) ? -3.0e38f : val;
      }
    }
    // online softmax update
    float mnew[8], alpha[8];
#pragma unroll
    for (int v = 0; v < 8; v++) {
      float mx = fmaxf(s[0][v], s[1][v]);
#pragma unroll
      for (int o = 8; o >= 1; o >>= 1) mx = fmaxf(mx, __shfl_xor(mx, o, 32));
      mnew[v] = fmaxf(mrow[v], mx);
      alpha[v] = __expf(mrow[v] - mnew[v]);
    }
#pragma unroll
    for (int v = 0; v < 8; v++) {
      float p0 = __expf(s[0][v] - mnew[v]);
      float p1 = __expf(s[1][v] - mnew[v]);
      float rs = p0 + p1;
#pragma unroll
      for (int o = 8; o >= 1; o >>= 1) rs += __shfl_xor(rs, o, 32);
      lrow[v] = lrow[v] * alpha[v] + rs;
      mrow[v] = mnew[v];
      int row = v + 8 * hf;
      pw[row * 32 + l15] = (bf16)p0;
      pw[row * 32 + 16 + l15] = (bf16)p1;
#pragma unroll
      for (int j = 0; j < 4; j++) oacc[j][v] *= alpha[v];
    }
    // P as A-fragment (16 q-rows x 32 keys), from LDS
    const bf16* pp = pw + l15 * 32 + hf * 16;
    v16bf pf = cat16(*(const v8bf*)pp, *(const v8bf*)(pp + 8));
    // O += P x V^T  (vT is (D, T) per head: standard NxK B-fragment)
#pragma unroll
    for (int j = 0; j < 4; j++) {
      const bf16* vb = vtbf + ((long)bh * D_ + j * 16 + l15) * T_ + c0 + hf * 16;
      v16bf bv = *(const v16bf*)vb;
      oacc[j] = wmma_bf16(pf, bv, oacc[j]);
    }
  }

  int h = bh % H_;
  int b = bh / H_;
#pragma unroll
  for (int j = 0; j < 4; j++) {
#pragma unroll
    for (int v = 0; v < 8; v++) {
      int row = v + 8 * hf;
      float val = oacc[j][v] / lrow[v];
      obf[(long)(b * T_ + q0 + row) * C_ + h * D_ + j * 16 + l15] = (bf16)val;
    }
  }
}

// ---------------------------------------------------------------- launch
extern "C" void kernel_launch(void* const* d_in, const int* in_sizes, int n_in,
                              void* d_out, int out_size, void* d_ws, size_t ws_size,
                              hipStream_t stream) {
  (void)in_sizes; (void)n_in; (void)out_size; (void)ws_size;
  const int* tokens = (const int*)d_in[0];
  const int* coords = (const int*)d_in[1];
  const int* ttype = (const int*)d_in[2];
  const float* pos_emb = (const float*)d_in[3];
  const float* feat_emb = (const float*)d_in[4];
  const float* qkv_w = (const float*)d_in[5];
  const float* out_w = (const float*)d_in[6];
  const float* q_scale = (const float*)d_in[7];
  const float* k_scale = (const float*)d_in[8];
  const float* ln1_w = (const float*)d_in[9];
  const float* ln1_b = (const float*)d_in[10];
  const float* ln2_w = (const float*)d_in[11];
  const float* ln2_b = (const float*)d_in[12];
  const float* fc1_w = (const float*)d_in[13];
  const float* fc2_w = (const float*)d_in[14];
  const float* res_a = (const float*)d_in[15];
  const float* res_m = (const float*)d_in[16];
  const float* skip_a = (const float*)d_in[17];
  const float* skip_m = (const float*)d_in[18];
  const float* lnf_w = (const float*)d_in[19];
  const float* lnf_b = (const float*)d_in[20];
  const float* pos_head = (const float*)d_in[21];
  const float* feat_head = (const float*)d_in[22];

  char* base = (char*)d_ws;
  size_t off = 0;
  auto alloc = [&](size_t bytes) -> void* {
    void* p = base + off;
    off = (off + bytes + 255) & ~(size_t)255;
    return p;
  };
  bf16* bqkvw = (bf16*)alloc((size_t)L_ * 3 * C_ * C_ * 2);
  bf16* boutw = (bf16*)alloc((size_t)L_ * C_ * C_ * 2);
  bf16* bfc1 = (bf16*)alloc((size_t)L_ * FF_ * C_ * 2);
  bf16* bfc2 = (bf16*)alloc((size_t)L_ * C_ * FF_ * 2);
  bf16* bph = (bf16*)alloc((size_t)VP_ * C_ * 2);
  bf16* bfh = (bf16*)alloc((size_t)VF_ * C_ * 2);
  float* x = (float*)alloc((size_t)M_ * C_ * 4);
  bf16* hbuf = (bf16*)alloc((size_t)M_ * C_ * 2);
  float* qkvbuf = (float*)alloc((size_t)M_ * 3 * C_ * 4);
  bf16* qb = (bf16*)alloc((size_t)B_ * H_ * T_ * D_ * 2);
  bf16* kb = (bf16*)alloc((size_t)B_ * H_ * T_ * D_ * 2);
  bf16* vtb = (bf16*)alloc((size_t)B_ * H_ * T_ * D_ * 2);
  bf16* ob = (bf16*)alloc((size_t)M_ * C_ * 2);
  bf16* mlpb = (bf16*)alloc((size_t)M_ * FF_ * 2);

  auto conv = [&](const float* src, bf16* dst, long n) {
    int blocks = (int)((n + 255) / 256);
    k_f32_to_bf16<<<blocks, 256, 0, stream>>>(src, dst, (int)n);
  };
  conv(qkv_w, bqkvw, (long)L_ * 3 * C_ * C_);
  conv(out_w, boutw, (long)L_ * C_ * C_);
  conv(fc1_w, bfc1, (long)L_ * FF_ * C_);
  conv(fc2_w, bfc2, (long)L_ * C_ * FF_);
  conv(pos_head, bph, (long)VP_ * C_);
  conv(feat_head, bfh, (long)VF_ * C_);

  auto gemm = [&](const bf16* A, const bf16* W, float* outf, bf16* outb,
                  const float* resid, const float* cr, const float* cs,
                  int M, int N, int K, int mode) {
    int nstrips = (N + 63) / 64;
    int tw = (M / 16) * nstrips;
    int blocks = (tw + 7) / 8;
    k_gemm<<<blocks, 256, 0, stream>>>(A, W, outf, outb, resid, cr, cs, M, N, K, mode);
  };

  k_embed<<<(M_ * C_) / 256, 256, 0, stream>>>(tokens, ttype, pos_emb, feat_emb, x);

  for (int l = 0; l < L_; l++) {
    k_ln<<<M_ / 8, 256, 0, stream>>>(x, ln1_w + l * C_, ln1_b + l * C_, hbuf);
    gemm(hbuf, bqkvw + (size_t)l * 3 * C_ * C_, qkvbuf, nullptr, nullptr, nullptr,
         nullptr, M_, 3 * C_, C_, 0);
    k_qkrope<<<(B_ * H_ * T_) / 8, 256, 0, stream>>>(qkvbuf, coords, q_scale + l * D_,
                                                     k_scale + l * D_, qb, kb, vtb);
    k_attn<<<(B_ * H_ * (T_ / 16)) / 8, 256, 0, stream>>>(qb, kb, vtb, ob);
    gemm(ob, boutw + (size_t)l * C_ * C_, x, nullptr, x, res_a + l, skip_a + l, M_, C_,
         C_, 1);
    k_ln<<<M_ / 8, 256, 0, stream>>>(x, ln2_w + l * C_, ln2_b + l * C_, hbuf);
    gemm(hbuf, bfc1 + (size_t)l * FF_ * C_, nullptr, mlpb, nullptr, nullptr, nullptr,
         M_, FF_, C_, 2);
    gemm(mlpb, bfc2 + (size_t)l * C_ * FF_, x, nullptr, x, res_m + l, skip_m + l, M_,
         C_, FF_, 1);
  }

  k_ln<<<M_ / 8, 256, 0, stream>>>(x, lnf_w, lnf_b, hbuf);
  float* outp = (float*)d_out;
  gemm(hbuf, bph, outp, nullptr, nullptr, nullptr, nullptr, M_, VP_, C_, 0);
  gemm(hbuf, bfh, outp + (size_t)M_ * VP_, nullptr, nullptr, nullptr, nullptr, M_, VF_,
       C_, 0);
}
